// PointFractalGenerator_75694503625222
// MI455X (gfx1250) — matrix operand order repaired and checked by reading, not compiled, not run on hardware
//
#include <hip/hip_runtime.h>
#include <cmath>

// ===========================================================================
// PointFractalGenerator forward — MI455X (gfx1250, wave32, WMMA).
//
// All 1x1 convs / linear layers run as token-major GEMMs on
// v_wmma_f32_16x16x32_f16. GEMM operands are staged in f16 (weights converted
// once per launch; activations written in f16 by producer epilogues), so the
// GEMM hot loop is pure b128-load -> v_wmma (branch-free: all fast GEMMs have
// N%32==0). The KNN mask makes attention exactly 16-sparse, so it is a gather
// kernel, not an NxN GEMM. All live activations (~130MB) fit in the 192MB L2.
// ===========================================================================

typedef _Float16 h16;
typedef __attribute__((ext_vector_type(16))) _Float16 v16h;
typedef __attribute__((ext_vector_type(8)))  _Float16 v8h;
typedef __attribute__((ext_vector_type(8)))  float    v8f;

#define NB    4      // batch
#define NPTS  2048   // points per cloud
#define TDIM  128    // DIM
#define UPF   2
#define NKNN  16

__device__ __forceinline__ float gelu_erf(float x) {
  return 0.5f * x * (1.0f + erff(x * 0.70710678118654752440f));
}

// ---------------------------------------------------------------------------
// f32 -> f16 conversion (weights pre-pass).
// ---------------------------------------------------------------------------
__global__ void __launch_bounds__(256)
cvt_f16_kernel(const float* __restrict__ src, h16* __restrict__ dst, int total)
{
  int t = (blockIdx.x * blockDim.x + threadIdx.x) * 2;
  if (t + 1 < total) {
    float2 v = *(const float2*)(src + t);
    dst[t] = (h16)v.x; dst[t + 1] = (h16)v.y;
  } else if (t < total) {
    dst[t] = (h16)src[t];
  }
}

// ---------------------------------------------------------------------------
// FAST WMMA GEMM (f16 operands):
//   C[M,N] = act(A[M,K] * W[N,K]^T + bias[N]) (+ res[M,N])
// Requires M%32==0, N%32==0, K%32==0 (true for every fast call site), which
// makes the inner loop branch-free. Layouts here (A/B k order contiguous, same for both):
//   A 16x32: lane l row M=l%16; half h -> K=16*(l/16)+h
//            => two contiguous 8-half (16B) runs at k0+16*hi and k0+16*hi+8
//   B 32x16: lane l col N=l%16; half h -> K=16*(l/16)+h => 16 contiguous (32B)
//   C 16x16: lane l, vgpr v -> M=v+8*(l/16), N=l%16
// One wave = 32x32 of C (2 A-frags x 2 B-frags = 4 WMMA / K-step);
// block = 4 waves = 32 rows x 128 cols. Hot loop: 8 b128 loads + 4 wmma.
// Epilogue can write f32 C and/or f16 Ch (stages next GEMM's operand).
// ---------------------------------------------------------------------------
__global__ void __launch_bounds__(128)
gemm_wmma_f16(const h16* __restrict__ A, const h16* __restrict__ W,
              const float* __restrict__ bias, const float* __restrict__ res,
              float* __restrict__ C, h16* __restrict__ Ch,
              int M, int N, int K, int act)
{
  const int lane = threadIdx.x & 31;
  const int wave = threadIdx.x >> 5;
  const int tn   = blockIdx.x * 128 + wave * 32;
  if (tn >= N) return;
  const int tm   = blockIdx.y * 32;
  const int hi   = lane >> 4;
  const int l16  = lane & 15;

  const h16* __restrict__ ar0 = A + (size_t)(tm + l16) * K + hi * 16;
  const h16* __restrict__ ar1 = ar0 + (size_t)16 * K;
  const h16* __restrict__ wr0 = W + (size_t)(tn + l16) * K + hi * 16;
  const h16* __restrict__ wr1 = wr0 + (size_t)16 * K;

  v8f acc00 = {}, acc01 = {}, acc10 = {}, acc11 = {};
  for (int k0 = 0; k0 < K; k0 += 32) {
    __builtin_prefetch(ar0 + k0 + 256, 0, 1);
    const v8h a0lo = *(const v8h*)(ar0 + k0);
    const v8h a0hi = *(const v8h*)(ar0 + k0 + 8);
    const v8h a1lo = *(const v8h*)(ar1 + k0);
    const v8h a1hi = *(const v8h*)(ar1 + k0 + 8);
    const v16h bv0 = *(const v16h*)(wr0 + k0);
    const v16h bv1 = *(const v16h*)(wr1 + k0);
    const v16h av0 = __builtin_shufflevector(a0lo, a0hi,
        0,1,2,3,4,5,6,7,8,9,10,11,12,13,14,15);
    const v16h av1 = __builtin_shufflevector(a1lo, a1hi,
        0,1,2,3,4,5,6,7,8,9,10,11,12,13,14,15);
    acc00 = __builtin_amdgcn_wmma_f32_16x16x32_f16(false, av0, false, bv0,
                                                   (short)0, acc00, false, false);
    acc10 = __builtin_amdgcn_wmma_f32_16x16x32_f16(false, av1, false, bv0,
                                                   (short)0, acc10, false, false);
    acc01 = __builtin_amdgcn_wmma_f32_16x16x32_f16(false, av0, false, bv1,
                                                   (short)0, acc01, false, false);
    acc11 = __builtin_amdgcn_wmma_f32_16x16x32_f16(false, av1, false, bv1,
                                                   (short)0, acc11, false, false);
  }

  const int bc0 = tn + l16;
  const int bc1 = bc0 + 16;
  const float bb0 = bias ? bias[bc0] : 0.0f;
  const float bb1 = bias ? bias[bc1] : 0.0f;
  auto emit = [&](int cm, int bc, float x) {
    if (act) x = gelu_erf(x);
    size_t off = (size_t)cm * N + bc;
    if (res) x += res[off];
    if (C)  C[off]  = x;
    if (Ch) Ch[off] = (h16)x;
  };
#pragma unroll
  for (int v = 0; v < 8; ++v) {
    const int cm0 = tm + v + 8 * hi;
    const int cm1 = cm0 + 16;
    emit(cm0, bc0, acc00[v] + bb0);
    emit(cm1, bc0, acc10[v] + bb0);
    emit(cm0, bc1, acc01[v] + bb1);
    emit(cm1, bc1, acc11[v] + bb1);
  }
}

static inline void gemm_f16(hipStream_t s, const h16* A, const h16* W,
                            const float* b, const float* res,
                            float* C, h16* Ch, int M, int N, int K, int act)
{
  dim3 g((N + 127) / 128, M / 32);
  gemm_wmma_f16<<<g, 128, 0, s>>>(A, W, b, res, C, Ch, M, N, K, act);
}

// ---------------------------------------------------------------------------
// SAFE WMMA GEMM (f32 operands, per-element guards) — only tiny edge GEMMs
// (conv_x K=3, conv_out N=3, global-feature path M=4).
// ---------------------------------------------------------------------------
__global__ void __launch_bounds__(128)
gemm_wmma_safe(const float* __restrict__ A, const float* __restrict__ W,
               const float* __restrict__ bias, const float* __restrict__ res,
               float* __restrict__ C, h16* __restrict__ Ch,
               int M, int N, int K, int act)
{
  const int lane = threadIdx.x & 31;
  const int wave = threadIdx.x >> 5;
  const int tn   = blockIdx.x * 64 + wave * 16;
  const int tm   = blockIdx.y * 16;
  const int hi   = lane >> 4;
  const int l16  = lane & 15;

  const int  arow  = tm + l16;
  const int  bcol  = tn + l16;
  const bool rowOK = arow < M;
  const bool colOK = bcol < N;
  const float* __restrict__ arp = A + (size_t)arow * K;
  const float* __restrict__ wrp = W + (size_t)bcol * K;

  v8f acc = {};
  for (int k0 = 0; k0 < K; k0 += 32) {
    v16h av, bv;
#pragma unroll
    for (int h = 0; h < 16; ++h) {
      int ka = k0 + hi * 16 + h;
      av[h] = (rowOK && ka < K) ? (h16)arp[ka] : (h16)0.0f;
      int kb = k0 + hi * 16 + h;
      bv[h] = (colOK && kb < K) ? (h16)wrp[kb] : (h16)0.0f;
    }
    acc = __builtin_amdgcn_wmma_f32_16x16x32_f16(false, av, false, bv,
                                                 (short)0, acc, false, false);
  }

  if (!colOK) return;
  const float bb = bias ? bias[bcol] : 0.0f;
#pragma unroll
  for (int v = 0; v < 8; ++v) {
    int cm = tm + v + 8 * hi;
    if (cm < M) {
      float x = acc[v] + bb;
      if (act) x = gelu_erf(x);
      size_t off = (size_t)cm * N + bcol;
      if (res) x += res[off];
      if (C)  C[off]  = x;
      if (Ch) Ch[off] = (h16)x;
    }
  }
}

static inline void gemm_safe(hipStream_t s, const float* A, const float* W,
                             const float* b, const float* res,
                             float* C, h16* Ch, int M, int N, int K, int act)
{
  dim3 g((N + 63) / 64, (M + 15) / 16);
  gemm_wmma_safe<<<g, 128, 0, s>>>(A, W, b, res, C, Ch, M, N, K, act);
}

// ---------------------------------------------------------------------------
// LayerNorm over last dim c, optional residual X+R; writes f32 Y and/or f16 Yh.
// One wave32 per token; cross-lane butterfly reductions.
// ---------------------------------------------------------------------------
__global__ void __launch_bounds__(256)
ln_kernel(const float* __restrict__ X, const float* __restrict__ R,
          const float* __restrict__ g, const float* __restrict__ bt,
          float* __restrict__ Y, h16* __restrict__ Yh, int tokens, int c)
{
  const int tok  = blockIdx.x * (blockDim.x >> 5) + (threadIdx.x >> 5);
  const int lane = threadIdx.x & 31;
  if (tok >= tokens) return;
  const float* x = X + (size_t)tok * c;
  const float* r = R ? R + (size_t)tok * c : nullptr;
  float s = 0.f, ss = 0.f;
  for (int i = lane; i < c; i += 32) {
    float v = x[i] + (r ? r[i] : 0.f);
    s += v; ss += v * v;
  }
#pragma unroll
  for (int o = 16; o; o >>= 1) {
    s  += __shfl_xor(s,  o, 32);
    ss += __shfl_xor(ss, o, 32);
  }
  const float m   = s / (float)c;
  const float var = ss / (float)c - m * m;
  const float inv = rsqrtf(var + 1e-5f);
  for (int i = lane; i < c; i += 32) {
    float v = x[i] + (r ? r[i] : 0.f);
    float y = (v - m) * inv * g[i] + bt[i];
    if (Y)  Y[(size_t)tok * c + i]  = y;
    if (Yh) Yh[(size_t)tok * c + i] = (h16)y;
  }
}

static inline void layernorm(hipStream_t s, const float* X, const float* R,
                             const float* g, const float* b,
                             float* Y, h16* Yh, int tokens, int c)
{
  ln_kernel<<<(tokens + 7) / 8, 256, 0, s>>>(X, R, g, b, Y, Yh, tokens, c);
}

// ---------------------------------------------------------------------------
// KNN: per query point, 16 nearest by squared distance (insertion sort scan).
// ---------------------------------------------------------------------------
__global__ void __launch_bounds__(256)
knn_kernel(const float* __restrict__ pt, int* __restrict__ idx, int b, int n)
{
  const int t = blockIdx.x * blockDim.x + threadIdx.x;
  if (t >= b * n) return;
  const int batch = t / n;
  const float* base = pt + (size_t)batch * n * 3;
  const float xi = pt[(size_t)t * 3 + 0];
  const float yi = pt[(size_t)t * 3 + 1];
  const float zi = pt[(size_t)t * 3 + 2];
  float bd[NKNN]; int bix[NKNN];
#pragma unroll
  for (int k = 0; k < NKNN; ++k) { bd[k] = 3.0e38f; bix[k] = 0; }
  for (int j = 0; j < n; ++j) {
    float dx = xi - base[(size_t)j * 3 + 0];
    float dy = yi - base[(size_t)j * 3 + 1];
    float dz = zi - base[(size_t)j * 3 + 2];
    float d  = dx * dx + dy * dy + dz * dz;
    if (d < bd[NKNN - 1]) {
      int p = NKNN - 1;
#pragma unroll
      for (int k = NKNN - 2; k >= 0; --k)
        if (d < bd[k]) { bd[k + 1] = bd[k]; bix[k + 1] = bix[k]; p = k; }
      bd[p] = d; bix[p] = j;
    }
  }
#pragma unroll
  for (int k = 0; k < NKNN; ++k) idx[(size_t)t * NKNN + k] = bix[k];
}

// ---------------------------------------------------------------------------
// Relative-position bias MLP: bias[t,k] = rp2( gelu( rp1(p_i - p_idx) ) )
// ---------------------------------------------------------------------------
__global__ void __launch_bounds__(256)
relpos_kernel(const float* __restrict__ pt, const int* __restrict__ idx,
              const float* __restrict__ W1, const float* __restrict__ b1,
              const float* __restrict__ W2, const float* __restrict__ b2,
              float* __restrict__ bias, int n, int c4, int total)
{
  const int t = blockIdx.x * blockDim.x + threadIdx.x;
  if (t >= total) return;
  const int k = t & (NKNN - 1);
  const int tok = t >> 4;
  const int batch = tok / n;
  const int j = idx[(size_t)tok * NKNN + k];
  const float* pi = pt + (size_t)tok * 3;
  const float* pj = pt + ((size_t)batch * n + j) * 3;
  const float r0 = pi[0] - pj[0], r1 = pi[1] - pj[1], r2 = pi[2] - pj[2];
  float acc = b2[0];
  for (int d = 0; d < c4; ++d) {
    float h = W1[d * 3 + 0] * r0 + W1[d * 3 + 1] * r1 + W1[d * 3 + 2] * r2 + b1[d];
    acc += gelu_erf(h) * W2[d];
  }
  bias[(size_t)tok * NKNN + k] = acc;
}

// ---------------------------------------------------------------------------
// 16-sparse multi-head attention. QKV packed f16 [tokens, 3c]; one block/token.
// Output O is f16 (it feeds the out-proj WMMA GEMM).
// ---------------------------------------------------------------------------
__global__ void __launch_bounds__(256)
attn_sparse_kernel(const h16* __restrict__ QKV, const int* __restrict__ idx,
                   const float* __restrict__ bias, h16* __restrict__ O,
                   int n, int c, float scale)
{
  const int t   = blockIdx.x;
  const int b   = t / n;
  const int tid = threadIdx.x;
  const int dh  = c >> 2;
  const int ld  = 3 * c;
  __shared__ int   sidx[NKNN];
  __shared__ float ssc[4][NKNN];
  __shared__ float sal[4][NKNN];
  if (tid < NKNN) sidx[tid] = idx[(size_t)t * NKNN + tid];
  __syncthreads();
  if (tid < 4 * NKNN) {
    const int h = tid >> 4, k = tid & (NKNN - 1);
    const h16* qp = QKV + (size_t)t * ld + h * dh;
    const h16* kp = QKV + ((size_t)(b * n + sidx[k])) * ld + c + h * dh;
    float s = 0.f;
    for (int d = 0; d < dh; ++d) s += (float)qp[d] * (float)kp[d];
    ssc[h][k] = s * scale + bias[(size_t)t * NKNN + k];
  }
  __syncthreads();
  if (tid < 4) {
    float mx = -3.0e38f;
#pragma unroll
    for (int k = 0; k < NKNN; ++k) mx = fmaxf(mx, ssc[tid][k]);
    float e[NKNN], den = 0.f;
#pragma unroll
    for (int k = 0; k < NKNN; ++k) { e[k] = expf(ssc[tid][k] - mx); den += e[k]; }
    const float inv = 1.f / den;
#pragma unroll
    for (int k = 0; k < NKNN; ++k) sal[tid][k] = e[k] * inv;
  }
  __syncthreads();
  for (int cc = tid; cc < c; cc += blockDim.x) {
    const int h = cc / dh;
    float o = 0.f;
#pragma unroll
    for (int k = 0; k < NKNN; ++k)
      o += sal[h][k] * (float)QKV[((size_t)(b * n + sidx[k])) * ld + 2 * c + cc];
    O[(size_t)t * c + cc] = (h16)o;
  }
}

// ---------------------------------------------------------------------------
// Layout plumbing kernels.
// ---------------------------------------------------------------------------
__global__ void transpose_pcd_kernel(const float* __restrict__ pcd,
                                     float* __restrict__ pt, int total)
{
  int t = blockIdx.x * blockDim.x + threadIdx.x;
  if (t >= total) return;
  int ch = t % 3; int tok = t / 3; int bb = tok / NPTS; int j = tok % NPTS;
  pt[t] = pcd[((size_t)bb * 3 + ch) * NPTS + j];
}

__global__ void concat_h0_kernel(const float* __restrict__ f1,
                                 const float* __restrict__ fg,
                                 h16* __restrict__ H, int total)
{
  int t = blockIdx.x * blockDim.x + threadIdx.x;
  if (t >= total) return;
  int ch = t & 255; int tok = t >> 8; int bb = tok / NPTS;
  H[t] = (h16)((ch < TDIM) ? f1[(size_t)tok * TDIM + ch]
                           : fg[(size_t)bb * TDIM + (ch - TDIM)]);
}

// P = conv_ps out [b*n, 256]; reshape [b,256,n]->[b,128,2n] means
// H3[b,c,m] = P[b*n + m%n][2c + (m>=n)]. Builds f16 Hc = concat(H3, tile(feat1))
// (consumed by conv_delta WMMA) and the f32 H3 network output.
__global__ void build_hc_h3_kernel(const float* __restrict__ P,
                                   const float* __restrict__ f1,
                                   h16* __restrict__ Hc,
                                   float* __restrict__ H3out, int total)
{
  int t = blockIdx.x * blockDim.x + threadIdx.x;
  if (t >= total) return;
  const int n2 = NPTS * UPF;
  int ch = t & 255; int tok = t >> 8; int m = tok % n2; int bb = tok / n2;
  int j = m % NPTS; int sft = (m >= NPTS) ? 1 : 0;
  float v;
  if (ch < TDIM) {
    v = P[((size_t)bb * NPTS + j) * (2 * TDIM) + 2 * ch + sft];
    H3out[((size_t)bb * TDIM + ch) * n2 + m] = v;
  } else {
    v = f1[((size_t)bb * NPTS + j) * TDIM + (ch - TDIM)];
  }
  Hc[(size_t)tok * (2 * TDIM) + ch] = (h16)v;
}

__global__ void finalize_pn_kernel(const float* __restrict__ pnt,
                                   const float* __restrict__ pcd,
                                   float* __restrict__ out, int total)
{
  int t = blockIdx.x * blockDim.x + threadIdx.x;
  if (t >= total) return;
  const int n2 = NPTS * UPF;
  int m = t % n2; int rest = t / n2; int ch = rest % 3; int bb = rest / 3;
  out[t] = pnt[((size_t)bb * n2 + m) * 3 + ch]
         + pcd[((size_t)bb * 3 + ch) * NPTS + (m % NPTS)];
}

// ---------------------------------------------------------------------------
// Host-side orchestration.
// ---------------------------------------------------------------------------
// Param flattening order (JAX pytree, sorted dict keys):
//  2:conv_1.W  3:conv_1.b  4:conv_11.W 5:conv_11.b  6:conv_delta.W 7:.b
//  8:conv_out.W 9:.b 10:conv_out1.W 11:.b 12:conv_ps.W 13:.b
// 14:conv_x.W 15:.b 16:conv_x1.W 17:.b
// SA block (base 18/36/54): +0 ff1.W +1 ff1.b +2 ff2.W +3 ff2.b
//  +4 ln12.b +5 ln12.g +6 ln13.b +7 ln13.g +8 out.W +9 out.b
//  +10 proj.W +11 proj.b +12 qkv.W +13 qkv.b +14 rp1.W +15 rp1.b +16 rp2.W +17 rp2.b
static void run_sa(hipStream_t s, void* const* d_in, h16* const* WH, int base,
                   const h16* Hinh, h16* Houth,
                   float* bufA, h16* bufAh, float* bufB, h16* bufBh,
                   float* bufC, h16* bufQKVh,
                   const float* pcdT, const int* idx, float* biasbuf,
                   int c, int din)
{
  const int T = NB * NPTS;
  auto P = [&](int off) { return (const float*)d_in[base + off]; };
  // x1 = proj(Hin) -> f32; q = LN13(x1) -> f32 (residual) + f16 (qkv operand)
  gemm_f16(s, Hinh, WH[base + 10], P(11), nullptr, bufA, nullptr, T, c, din, 0);
  layernorm(s, bufA, nullptr, P(7), P(6), bufA, bufAh, T, c);
  // KNN relative-position bias
  relpos_kernel<<<(T * NKNN + 255) / 256, 256, 0, s>>>(
      pcdT, idx, P(14), P(15), P(16), P(17), biasbuf, NPTS, c / 4, T * NKNN);
  // QKV packed [T, 3c] — f16 only (consumed by gather attention)
  gemm_f16(s, bufAh, WH[base + 12], P(13), nullptr, nullptr, bufQKVh, T, 3 * c, c, 0);
  // 16-sparse attention -> f16 O
  attn_sparse_kernel<<<T, 256, 0, s>>>(bufQKVh, idx, biasbuf, bufBh, NPTS, c,
                                       1.0f / sqrtf((float)(c / 4)));
  // out proj -> f32; x = LN12(q + o) -> f32 (ff2 residual) + f16 (ff1 operand)
  gemm_f16(s, bufBh, WH[base + 8], P(9), nullptr, bufC, nullptr, T, c, c, 0);
  layernorm(s, bufA, bufC, P(5), P(4), bufB, bufBh, T, c);
  // FFN: ff1 (gelu, f16 only) ; ff2 + residual -> f16 Hout (next proj operand)
  gemm_f16(s, bufBh, WH[base + 0], P(1), nullptr, nullptr, bufQKVh, T, 1024, c, 1);
  gemm_f16(s, bufQKVh, WH[base + 2], P(3), bufB, nullptr, Houth, T, c, 1024, 0);
}

extern "C" void kernel_launch(void* const* d_in, const int* in_sizes, int n_in,
                              void* d_out, int out_size, void* d_ws, size_t ws_size,
                              hipStream_t stream)
{
  (void)in_sizes; (void)n_in; (void)out_size; (void)ws_size;
  const float* pcd  = (const float*)d_in[0];
  const float* gfe  = (const float*)d_in[1];   // [4,512,1] == token-major [4,512]
  auto P = [&](int i) { return (const float*)d_in[i]; };

  const int T  = NB * NPTS;       // 8192
  const int T2 = NB * NPTS * UPF; // 16384

  // Workspace arena (256B-aligned slots)
  char* wsb = (char*)d_ws;
  size_t off = 0;
  auto allocB = [&](size_t bytes) {
    void* p = wsb + off; off = (off + bytes + 255) & ~(size_t)255; return p;
  };
  // f32 buffers
  float* pcdT    = (float*)allocB(sizeof(float) * (size_t)T * 3);
  float* feat1   = (float*)allocB(sizeof(float) * (size_t)T * TDIM);
  float* g256    = (float*)allocB(sizeof(float) * (size_t)NB * 256);
  float* fgT     = (float*)allocB(sizeof(float) * (size_t)NB * TDIM);
  float* biasbuf = (float*)allocB(sizeof(float) * (size_t)T * NKNN);
  float* bufA    = (float*)allocB(sizeof(float) * (size_t)T * 512);  // x1/q; later conv_ps out P
  float* bufB    = (float*)allocB(sizeof(float) * (size_t)T * 512);  // x (ff2 residual)
  float* bufC    = (float*)allocB(sizeof(float) * (size_t)T * 512);  // out-proj; later conv_out1 out
  float* pnT     = (float*)allocB(sizeof(float) * (size_t)T2 * 3);
  int*   idxbuf  = (int*)  allocB(sizeof(int)   * (size_t)T * NKNN);
  // f16 buffers
  h16* t64h    = (h16*)allocB(sizeof(h16) * (size_t)T * 64);
  h16* Hcurh   = (h16*)allocB(sizeof(h16) * (size_t)T * 512);
  h16* Hnexth  = (h16*)allocB(sizeof(h16) * (size_t)T * 512);
  h16* bufAh   = (h16*)allocB(sizeof(h16) * (size_t)T * 512);
  h16* bufBh   = (h16*)allocB(sizeof(h16) * (size_t)T * 512);   // attn O / x / H4
  h16* bufQKVh = (h16*)allocB(sizeof(h16) * (size_t)T * 1536);  // qkv / ff1 / Hc
  h16* wslab   = (h16*)allocB(sizeof(h16) * (size_t)7 * 1024 * 1024);

  // --- convert all WMMA weights to f16 once ---
  h16* WH[72] = {};
  size_t woff = 0;
  auto wcvt = [&](int i, size_t n) {
    h16* d = wslab + woff; woff += (n + 127) & ~(size_t)127;
    cvt_f16_kernel<<<(int)(((n + 1) / 2 + 255) / 256), 256, 0, stream>>>(P(i), d, (int)n);
    WH[i] = d;
  };
  wcvt(16, 128 * 64);     // conv_x1
  wcvt(12, 256 * 256);    // conv_ps
  wcvt(6,  128 * 256);    // conv_delta
  wcvt(10, 64 * 128);     // conv_out1
  {
    const int cs[3]  = {512, 512, 256};
    const int dins[3] = {256, 512, 512};
    const int bases[3] = {18, 36, 54};
    for (int s3 = 0; s3 < 3; ++s3) {
      int c = cs[s3], din = dins[s3], b = bases[s3];
      wcvt(b + 0,  (size_t)1024 * c);   // ff1
      wcvt(b + 2,  (size_t)c * 1024);   // ff2
      wcvt(b + 8,  (size_t)c * c);      // out
      wcvt(b + 10, (size_t)c * din);    // proj
      wcvt(b + 12, (size_t)3 * c * c);  // qkv
    }
  }

  // --- geometry prep ---
  transpose_pcd_kernel<<<(T * 3 + 255) / 256, 256, 0, stream>>>(pcd, pcdT, T * 3);
  knn_kernel<<<(T + 255) / 256, 256, 0, stream>>>(pcdT, idxbuf, NB, NPTS);

  // --- feat1 = conv_x1(gelu(conv_x(pcd))) ---
  gemm_safe(stream, pcdT, P(14), P(15), nullptr, nullptr, t64h, T, 64, 3, 1);
  gemm_f16(stream, t64h, WH[16], P(17), nullptr, feat1, nullptr, T, TDIM, 64, 0);

  // --- fg = conv_1(gelu(conv_11(gfeat))) ---
  gemm_safe(stream, gfe,  P(4), P(5), nullptr, g256, nullptr, NB, 256,  512, 1);
  gemm_safe(stream, g256, P(2), P(3), nullptr, fgT,  nullptr, NB, TDIM, 256, 0);

  // --- H0 = concat(feat1, broadcast fg) -> f16 (proj operand) ---
  concat_h0_kernel<<<(T * 256 + 255) / 256, 256, 0, stream>>>(feat1, fgT, Hcurh, T * 256);

  // --- 3x PointSparseAttn ---
  run_sa(stream, d_in, WH, 18, Hcurh,  Hnexth, bufA, bufAh, bufB, bufBh, bufC,
         bufQKVh, pcdT, idxbuf, biasbuf, 512, 256);
  run_sa(stream, d_in, WH, 36, Hnexth, Hcurh,  bufA, bufAh, bufB, bufBh, bufC,
         bufQKVh, pcdT, idxbuf, biasbuf, 512, 512);
  run_sa(stream, d_in, WH, 54, Hcurh,  Hnexth, bufA, bufAh, bufB, bufBh, bufC,
         bufQKVh, pcdT, idxbuf, biasbuf, 256, 512);

  // --- pixel shuffle head ---
  float* out_pn = (float*)d_out;
  float* out_H3 = out_pn + (size_t)NB * 3 * NPTS * UPF;
  gemm_f16(stream, Hnexth, WH[12], P(13), nullptr, bufA, nullptr,
           T, 2 * TDIM, 2 * TDIM, 0);                                   // conv_ps -> P (f32)
  build_hc_h3_kernel<<<(T2 * 256 + 255) / 256, 256, 0, stream>>>(
      bufA, feat1, bufQKVh, out_H3, T2 * 256);                          // Hc (f16) + H3 out
  gemm_f16(stream, bufQKVh, WH[6], P(7), nullptr, nullptr, bufBh,
           T2, TDIM, 2 * TDIM, 0);                                      // conv_delta -> f16
  gemm_f16(stream, bufBh, WH[10], P(11), nullptr, bufC, nullptr,
           T2, 64, TDIM, 1);                                            // conv_out1+gelu -> f32
  gemm_safe(stream, bufC, P(8), P(9), nullptr, pnT, nullptr, T2, 3, 64, 0); // conv_out
  finalize_pn_kernel<<<(NB * 3 * NPTS * UPF + 255) / 256, 256, 0, stream>>>(
      pnT, pcd, out_pn, NB * 3 * NPTS * UPF);
}
